// MultiHeadAttention_56152402428581
// MI455X (gfx1250) — matrix operand order repaired and checked by reading, not compiled, not run on hardware
//
#include <hip/hip_runtime.h>
#include <hip/hip_fp16.h>

#define B_  4
#define S_  2048
#define D_  1024
#define H_  16
#define DH_ 64
#define NW_ (S_ / 32)          // mask words per row (64)

static constexpr float LN_EPS  = 1e-5f;
static constexpr float SCALE   = 0.03125f;   // 1/sqrt(D) = 1/32
static constexpr float MASKVAL = -1e-10f;    // reference uses -1e-10, NOT -inf

typedef __attribute__((ext_vector_type(16))) _Float16     v16h;
typedef __attribute__((ext_vector_type(8)))  _Float16     v8h;
typedef __attribute__((ext_vector_type(4)))  _Float16     v4h;
typedef __attribute__((ext_vector_type(8)))  float        v8f;
typedef __attribute__((ext_vector_type(4)))  float        v4f;
typedef __attribute__((ext_vector_type(4)))  int          v4i;

// ---------------------------------------------------------------------------
// Pass 1a: Kh[b,s,d] = f16(q*projK)  (row-major, WMMA-B friendly)
//          VT[b,h,dh,s] = f16(q*projV) (transposed, WMMA-B friendly for P*V)
// One block per (b,s) row; q reads and Kh writes fully coalesced.
// ---------------------------------------------------------------------------
__launch_bounds__(256)
__global__ void prep_kv(const float* __restrict__ q,
                        const float* __restrict__ projK,
                        const float* __restrict__ projV,
                        _Float16* __restrict__ Kh,
                        _Float16* __restrict__ VT)
{
    const int row = blockIdx.x;            // b*S + s
    const int b   = row / S_;
    const int s   = row % S_;
    const int d0  = threadIdx.x * 4;
    const v4f qv = *(const v4f*)(q + (size_t)row * D_ + d0);
    const v4f pk = *(const v4f*)(projK + d0);
    const v4f pv = *(const v4f*)(projV + d0);
    v4h kh;
#pragma unroll
    for (int i = 0; i < 4; ++i) kh[i] = (_Float16)(qv[i] * pk[i]);
    *(v4h*)(Kh + (size_t)row * D_ + d0) = kh;
    const int hh = d0 >> 6;
    const int dh = d0 & 63;
    _Float16* vdst = VT + ((size_t)(b * H_ + hh) * DH_ + dh) * S_ + s;
#pragma unroll
    for (int i = 0; i < 4; ++i) vdst[(size_t)i * S_] = (_Float16)(qv[i] * pv[i]);
}

// ---------------------------------------------------------------------------
// Pass 1b: bit-pack masks: Mb[(m*S + row)*NW + w] holds keys w*32..w*32+31.
// ---------------------------------------------------------------------------
__launch_bounds__(256)
__global__ void prep_mask(const int* __restrict__ masks, unsigned* __restrict__ Mb)
{
    const size_t w = (size_t)blockIdx.x * 256 + threadIdx.x;
    const int* src = masks + w * 32;
    unsigned bits = 0;
#pragma unroll
    for (int g = 0; g < 8; ++g) {
        const v4i m = *(const v4i*)(src + g * 4);
        bits |= ((unsigned)(m[0] & 1) << (g * 4))     |
                ((unsigned)(m[1] & 1) << (g * 4 + 1)) |
                ((unsigned)(m[2] & 1) << (g * 4 + 2)) |
                ((unsigned)(m[3] & 1) << (g * 4 + 3));
    }
    Mb[w] = bits;
}

// ---------------------------------------------------------------------------
// Pass 2: barrier-free flash attention.  8 independent waves per block, each
// owning a 16-row query tile of one (b,h).  Per 64-key iteration:
//   8 wmma (QK^T) + 2 wmma (row-sums via P*ones) + 8 wmma (P*V),
// with K/V B-fragments loaded straight from L2-resident f16 workspace.
// ---------------------------------------------------------------------------
__launch_bounds__(256)
__global__ void attn_kernel(const float*    __restrict__ q,
                            const float*    __restrict__ projQ,
                            const _Float16* __restrict__ Kh,
                            const _Float16* __restrict__ VT,
                            const unsigned* __restrict__ Mb,
                            float*          __restrict__ ctx)   // d_out
{
    __shared__ __align__(32) _Float16 Plds[8][16 * 64];  // per-wave P transpose tile

    const int tid  = threadIdx.x;
    const int wave = tid >> 5;
    const int lane = tid & 31;
    const int half = lane >> 4;
    const int ln   = lane & 15;

    const int nQB  = S_ / 128;
    const int qblk = blockIdx.x % nQB;
    const int bh   = blockIdx.x / nQB;
    const int b    = bh / H_;
    const int h    = bh % H_;
    const int c0   = h * DH_;
    const int qbase = qblk * 128 + wave * 16;

    const _Float16* khb = Kh + (size_t)b * S_ * D_ + c0;            // [key][dh]
    const _Float16* vtb = VT + (size_t)(b * H_ + h) * DH_ * S_;     // [dh][key]
    // head-major mask tiling bug: mask index = (b*H + h) % B = h % B
    const unsigned* mbb = Mb + (size_t)(h % B_) * S_ * NW_;

    // ---- persistent Q A-fragments (A 16x32 f16, contiguous k = 16*half + i), q*projQ ----
    v16h qa[2];
    {
        const float* qrow = q + ((size_t)b * S_ + qbase + ln) * D_ + c0;
        const float* pQ   = projQ + c0;
#pragma unroll
        for (int c = 0; c < 2; ++c) {
            const int b0 = c * 32 + half * 16;
#pragma unroll
            for (int i = 0; i < 8; ++i) {
                qa[c][i]     = (_Float16)(qrow[b0 + i]      * pQ[b0 + i]);
                qa[c][8 + i] = (_Float16)(qrow[b0 + 8 + i] * pQ[b0 + 8 + i]);
            }
        }
    }

    const v8f vzero = {0.f,0.f,0.f,0.f,0.f,0.f,0.f,0.f};
    v8f O[4];
#pragma unroll
    for (int t = 0; t < 4; ++t) O[t] = vzero;
    float mrow[8], lrow[8];
#pragma unroll
    for (int r = 0; r < 8; ++r) { mrow[r] = -3.0e38f; lrow[r] = 0.f; }

    v16h ones;
#pragma unroll
    for (int i = 0; i < 16; ++i) ones[i] = (_Float16)1.0f;

    const unsigned bm0 = 1u << ln;        // bit for subtiles 0/2
    const unsigned bm1 = bm0 << 16;       // bit for subtiles 1/3

    for (int kb = 0; kb < S_; kb += 64) {
        if (kb + 64 < S_) {               // gfx1250 global_prefetch_b8, next block
            __builtin_prefetch(khb + (size_t)(kb + 64 + lane) * D_, 0, 3);
            __builtin_prefetch(khb + (size_t)(kb + 96 + lane) * D_, 0, 3);
            __builtin_prefetch(vtb + (size_t)lane        * S_ + kb + 64, 0, 3);
            __builtin_prefetch(vtb + (size_t)(lane + 32) * S_ + kb + 64, 0, 3);
        }

        // ---- mask words for the 64-key block (2 words per query row) ----
        unsigned mw0[8], mw1[8];
#pragma unroll
        for (int r = 0; r < 8; ++r) {
            const unsigned* mp = mbb + (size_t)(qbase + r + half * 8) * NW_ + (kb >> 5);
            mw0[r] = mp[0]; mw1[r] = mp[1];
        }

        // ---- scores: four 16x16 N-tiles, B-frags straight from global ----
        float sv[4][8];
#pragma unroll
        for (int t = 0; t < 4; ++t) {
            const _Float16* kr = khb + (size_t)(kb + t * 16 + ln) * D_;
            const v16h kf0 = *(const v16h*)(kr + half * 16);        // dh 0..31
            const v16h kf1 = *(const v16h*)(kr + 32 + half * 16);   // dh 32..63
            v8f acc = vzero;
            acc = __builtin_amdgcn_wmma_f32_16x16x32_f16(false, qa[0], false, kf0,
                                                         (short)0, acc, false, false);
            acc = __builtin_amdgcn_wmma_f32_16x16x32_f16(false, qa[1], false, kf1,
                                                         (short)0, acc, false, false);
            const unsigned bm = (t & 1) ? bm1 : bm0;
#pragma unroll
            for (int r = 0; r < 8; ++r) {
                float s = acc[r] * SCALE;
                const unsigned w = (t < 2) ? mw0[r] : mw1[r];
                if (w & bm) s = MASKVAL;
                sv[t][r] = s;
            }
        }

        // ---- online softmax: shuffle max, exp; row-sums come from WMMA ----
        float alpha[8], pv[4][8];
#pragma unroll
        for (int r = 0; r < 8; ++r) {
            float mx = fmaxf(fmaxf(sv[0][r], sv[1][r]), fmaxf(sv[2][r], sv[3][r]));
#pragma unroll
            for (int off = 8; off >= 1; off >>= 1)
                mx = fmaxf(mx, __shfl_xor(mx, off, 32));
            const float nm = fmaxf(mrow[r], mx);
            alpha[r] = __expf(mrow[r] - nm);
            mrow[r]  = nm;
#pragma unroll
            for (int t = 0; t < 4; ++t) pv[t][r] = __expf(sv[t][r] - nm);
        }

        // ---- re-layout P (C-layout -> A-fragments) via per-wave LDS ----
        _Float16* P = Plds[wave];
#pragma unroll
        for (int t = 0; t < 4; ++t)
#pragma unroll
            for (int r = 0; r < 8; ++r)
                P[(r + half * 8) * 64 + t * 16 + ln] = (_Float16)pv[t][r];
        asm volatile("s_wait_dscnt 0" ::: "memory");
        v16h pa[2];
#pragma unroll
        for (int c = 0; c < 2; ++c) {
            const v8h lo = *(const v8h*)&P[ln * 64 + c * 32 + half * 16];
            const v8h hi = *(const v8h*)&P[ln * 64 + c * 32 + half * 16 + 8];
#pragma unroll
            for (int i = 0; i < 8; ++i) { pa[c][i] = lo[i]; pa[c][8 + i] = hi[i]; }
        }

        // ---- row sums at f16-P precision: Csum = P * ones (all columns equal) ----
        v8f csum = vzero;
        csum = __builtin_amdgcn_wmma_f32_16x16x32_f16(false, pa[0], false, ones,
                                                      (short)0, csum, false, false);
        csum = __builtin_amdgcn_wmma_f32_16x16x32_f16(false, pa[1], false, ones,
                                                      (short)0, csum, false, false);
#pragma unroll
        for (int r = 0; r < 8; ++r) lrow[r] = lrow[r] * alpha[r] + csum[r];

        // ---- rescale O, then O += P(16x64) x V(64x64-slice) ----
#pragma unroll
        for (int t = 0; t < 4; ++t)
#pragma unroll
            for (int r = 0; r < 8; ++r) O[t][r] *= alpha[r];
#pragma unroll
        for (int t = 0; t < 4; ++t) {
            const _Float16* vr = vtb + (size_t)(t * 16 + ln) * S_ + kb;
            const v16h vf0 = *(const v16h*)(vr + half * 16);        // keys 0..31
            const v16h vf1 = *(const v16h*)(vr + 32 + half * 16);   // keys 32..63
            O[t] = __builtin_amdgcn_wmma_f32_16x16x32_f16(false, pa[0], false, vf0,
                                                          (short)0, O[t], false, false);
            O[t] = __builtin_amdgcn_wmma_f32_16x16x32_f16(false, pa[1], false, vf1,
                                                          (short)0, O[t], false, false);
        }
    }

    // ---- normalize and write ctx ----
#pragma unroll
    for (int t = 0; t < 4; ++t)
#pragma unroll
        for (int r = 0; r < 8; ++r) {
            const int row = qbase + r + half * 8;
            ctx[((size_t)b * S_ + row) * D_ + c0 + t * 16 + ln] = O[t][r] / lrow[r];
        }
}

// ---------------------------------------------------------------------------
// Pass 3: residual + LayerNorm, in place on d_out.  One block per row.
// ---------------------------------------------------------------------------
__launch_bounds__(256)
__global__ void ln_kernel(const float* __restrict__ q,
                          const float* __restrict__ gamma,
                          const float* __restrict__ beta,
                          float*       __restrict__ out)
{
    __shared__ float ws1[8], ws2[8];
    const int row = blockIdx.x;
    const int tid = threadIdx.x;
    const float* qr = q   + (size_t)row * D_;
    float*     orow = out + (size_t)row * D_;

    float x[4], s = 0.f, s2 = 0.f;
#pragma unroll
    for (int i = 0; i < 4; ++i) {
        const int idx = tid * 4 + i;
        x[i] = orow[idx] + qr[idx];
        s += x[i]; s2 += x[i] * x[i];
    }
#pragma unroll
    for (int off = 16; off >= 1; off >>= 1) {
        s  += __shfl_xor(s,  off, 32);
        s2 += __shfl_xor(s2, off, 32);
    }
    if ((tid & 31) == 0) { ws1[tid >> 5] = s; ws2[tid >> 5] = s2; }
    __syncthreads();
    float ts = 0.f, ts2 = 0.f;
#pragma unroll
    for (int w = 0; w < 8; ++w) { ts += ws1[w]; ts2 += ws2[w]; }
    const float mean = ts * (1.0f / D_);
    const float var  = ts2 * (1.0f / D_) - mean * mean;
    const float inv  = rsqrtf(var + LN_EPS);
#pragma unroll
    for (int i = 0; i < 4; ++i) {
        const int idx = tid * 4 + i;
        orow[idx] = (x[i] - mean) * inv * gamma[idx] + beta[idx];
    }
}

extern "C" void kernel_launch(void* const* d_in, const int* in_sizes, int n_in,
                              void* d_out, int out_size, void* d_ws, size_t ws_size,
                              hipStream_t stream) {
    (void)in_sizes; (void)n_in; (void)out_size; (void)ws_size;
    const float* q     = (const float*)d_in[0];
    // d_in[1] (k) and d_in[2] (v) are dead per the reference's preserved bug.
    const int*   masks = (const int*)  d_in[3];
    const float* projQ = (const float*)d_in[4];
    const float* projK = (const float*)d_in[5];
    const float* projV = (const float*)d_in[6];
    const float* gamma = (const float*)d_in[7];
    const float* beta  = (const float*)d_in[8];
    float* out = (float*)d_out;

    char* ws = (char*)d_ws;
    _Float16* Kh = (_Float16*)ws;                                        // 16 MB
    _Float16* VT = (_Float16*)(ws + (size_t)B_ * S_ * D_ * 2);           // 16 MB
    unsigned* Mb = (unsigned*)(ws + (size_t)B_ * S_ * D_ * 4);           //  2 MB

    prep_kv  <<<dim3(B_ * S_),               256, 0, stream>>>(q, projK, projV, Kh, VT);
    prep_mask<<<dim3((B_ * S_ * NW_) / 256), 256, 0, stream>>>(masks, Mb);
    attn_kernel<<<dim3(B_ * H_ * (S_ / 128)), 256, 0, stream>>>(q, projQ, Kh, VT, Mb, out);
    ln_kernel<<<dim3(B_ * S_),               256, 0, stream>>>(q, gamma, beta, out);
}
